// CrossQuant_90074054132057
// MI455X (gfx1250) — matrix operand order repaired and checked by reading, not compiled, not run on hardware
//
#include <hip/hip_runtime.h>
#include <math.h>

// Problem constants (match reference setup_inputs)
#define B_  4096
#define D_  128
#define M_  8192
#define K_  4

typedef __attribute__((ext_vector_type(16))) _Float16 v16h;
typedef __attribute__((ext_vector_type(8)))  _Float16 v8h;
typedef __attribute__((ext_vector_type(8)))  float    v8f;

union HF16 { v16h v; v8h h[2]; };

// Order-preserving float <-> uint encoding (branchless)
__device__ __forceinline__ unsigned fenc(float f) {
  unsigned u = __float_as_uint(f);
  return u ^ ((unsigned)((int)u >> 31) | 0x80000000u);
}
__device__ __forceinline__ float fdec(unsigned e) {
  return (e & 0x80000000u) ? __uint_as_float(e & 0x7FFFFFFFu)
                           : __uint_as_float(~e);
}

// ---------------------------------------------------------------------------
// init: residuals = inputs (f32 + f16 mirror), quantized sums = 0
__global__ void init_main_kernel(const float* __restrict__ pcf, const float* __restrict__ plm,
                                 float* res_p, float* res_l,
                                 _Float16* resh_p, _Float16* resh_l,
                                 float* qp, float* ql) {
  size_t i = (size_t)blockIdx.x * 256 + threadIdx.x;
  if (i >= (size_t)B_ * D_) return;
  float a = pcf[i], b = plm[i];
  res_p[i] = a; resh_p[i] = (_Float16)a; qp[i] = 0.f;
  res_l[i] = b; resh_l[i] = (_Float16)b; ql[i] = 0.f;
}

__global__ void init_misc_kernel(float* counts, float* scal) {
  size_t i = (size_t)blockIdx.x * 256 + threadIdx.x;
  if (i < (size_t)K_ * 2 * M_) counts[i] = 0.f;
  if (i < 16) scal[i] = 0.f;   // [0]=lsum [1..4]=commit sums [8]=max-enc (0 < any fenc)
}

// ---------------------------------------------------------------------------
// Convert codebooks to f16 + compute ||E||^2 per row. One wave32 per row.
__global__ void eprep_kernel(const float* __restrict__ emb, _Float16* __restrict__ Eh,
                             float* __restrict__ enorm) {
  const int lane = threadIdx.x & 31, wv = threadIdx.x >> 5;
  const int row = blockIdx.x * 8 + wv;
  const float* er = emb + (size_t)row * D_;
  _Float16* eh = Eh + (size_t)row * D_;
  float s = 0.f;
#pragma unroll
  for (int t = 0; t < 4; ++t) {
    int d = lane + t * 32;
    float e = er[d];
    eh[d] = (_Float16)e;
    s += e * e;
  }
#pragma unroll
  for (int m = 16; m; m >>= 1) s += __shfl_xor(s, m, 32);
  if (lane == 0) enorm[row] = s;
}

// ---------------------------------------------------------------------------
// Fragment load + WMMA/argmin step helpers for the distance kernel.
// B 32x16 layout: lanes 0-15 hold K {8h@0, 8h@16}; lanes 16-31 {8h@8, 8h@24}.
__device__ __forceinline__ void load_bfrag(HF16* b, const _Float16* __restrict__ ep, int half) {
#pragma unroll
  for (int kk = 0; kk < 4; ++kk) {
    b[kk].h[0] = *(const v8h*)(ep + kk * 32 + half * 8);
    b[kk].h[1] = *(const v8h*)(ep + kk * 32 + 16 + half * 8);
  }
}

// key packing: top 19 bits = monotone float encoding, low 13 bits = index.
__device__ __forceinline__ void argmin_step(const HF16* __restrict__ a,
                                            const HF16* __restrict__ b,
                                            float en, int ncol,
                                            unsigned* __restrict__ best) {
  v8f c = {};
#pragma unroll
  for (int kk = 0; kk < 4; ++kk)
    c = __builtin_amdgcn_wmma_f32_16x16x32_f16(false, a[kk].v, false, b[kk].v,
                                               (short)0, c, false, false);
#pragma unroll
  for (int r = 0; r < 8; ++r) {
    float dist = en - 2.0f * c[r];                  // row = r + 8*half, col = ncol
    unsigned p = (fenc(dist) & 0xFFFFE000u) | (unsigned)ncol;
    best[r] = best[r] < p ? best[r] : p;            // v_min_u32
  }
}

// WMMA distance + fused argmin. Block = 4 waves, owns 16 rows, sweeps all M.
// dist ordering key: enorm[m] - 2 * (res . E[m])   (||res||^2 constant/row)
// Ping-pong B fragments (no rotation moves); per-lane running minima in
// registers; one ds_min_u32 per (lane,row) at the very end.
__global__ __launch_bounds__(128) void argmin_kernel(
    const _Float16* __restrict__ resh, const _Float16* __restrict__ Eh,
    const float* __restrict__ enorm, int kstage, int* __restrict__ amin) {
  __shared__ unsigned slots[16];
  const int tid = threadIdx.x;
  const int lane = tid & 31, wv = tid >> 5;
  const int half = lane >> 4, lr = lane & 15;
  const int r0 = blockIdx.x * 16;
  if (tid < 16) slots[tid] = 0xFFFFFFFFu;
  __syncthreads();

  // A fragments: 16 residual rows x K=128, hoisted out of the M loop.
  // 16-bit A layout: lanes 0-15 hold K {8h@0, 8h@16}; lanes 16-31 {8h@8, 8h@24}.
  HF16 a[4];
  {
    const _Float16* ap = resh + (size_t)(r0 + lr) * D_;
#pragma unroll
    for (int kk = 0; kk < 4; ++kk) {
      const _Float16* p = ap + kk * 32;
      a[kk].h[0] = *(const v8h*)(p + half * 8);
      a[kk].h[1] = *(const v8h*)(p + 16 + half * 8);
    }
  }

  unsigned best[8];
#pragma unroll
  for (int r = 0; r < 8; ++r) best[r] = 0xFFFFFFFFu;

  const size_t ebase = (size_t)kstage * M_;
  int ncol = wv * 16 + lr;
  HF16 b0[4], b1[4];
  float en0, en1;
  load_bfrag(b0, Eh + (ebase + ncol) * (size_t)D_, half);
  en0 = enorm[ebase + ncol];

#pragma unroll 1
  for (int it = 0; it < M_ / 128; ++it) {
    const int ncol1 = ncol + 64;                    // always < M_
    load_bfrag(b1, Eh + (ebase + ncol1) * (size_t)D_, half);
    en1 = enorm[ebase + ncol1];
    argmin_step(a, b0, en0, ncol, best);

    const int ncol2 = ncol + 128;
    if (it + 1 < M_ / 128) {
      load_bfrag(b0, Eh + (ebase + ncol2) * (size_t)D_, half);
      en0 = enorm[ebase + ncol2];
    }
    argmin_step(a, b1, en1, ncol1, best);
    ncol = ncol2;
  }

#pragma unroll
  for (int r = 0; r < 8; ++r)
    atomicMin(&slots[r + 8 * half], best[r]);       // ds_min_u32, once per lane
  __syncthreads();
  if (tid < 16) amin[r0 + tid] = (int)(slots[tid] & 0x1FFFu);
}

// ---------------------------------------------------------------------------
// Gather chosen embedding, update residual (+f16 mirror), accumulate q,
// histogram counts, emit sem_ids (as float). One wave per row.
__global__ void update_kernel(const float* __restrict__ emb, const int* __restrict__ amin,
                              float* __restrict__ res, _Float16* __restrict__ resh,
                              float* __restrict__ q, float* __restrict__ cnt,
                              float* __restrict__ out_sem, int kstage) {
  const int lane = threadIdx.x & 31, wv = threadIdx.x >> 5;
  const int b = blockIdx.x * 8 + wv;
  const int idx = amin[b];
  const float* e = emb + ((size_t)kstage * M_ + idx) * (size_t)D_;
  const size_t rb = (size_t)b * D_;
#pragma unroll
  for (int t = 0; t < 4; ++t) {
    int d = lane + t * 32;
    float ev = e[d];
    q[rb + d] += ev;
    float nr = res[rb + d] - ev;
    res[rb + d] = nr;
    resh[rb + d] = (_Float16)nr;
  }
  if (lane == 0) {
    atomicAdd(&cnt[idx], 1.0f);
    out_sem[(size_t)b * K_ + kstage] = (float)idx;
  }
}

// ---------------------------------------------------------------------------
// perplexity = exp(-sum p log(p+1e-10)), one block per (k,s)
__global__ void perp_kernel(const float* __restrict__ counts, float* __restrict__ out_perp) {
  __shared__ float red[256];
  const int ks = blockIdx.x;
  const float* c = counts + (size_t)ks * M_;
  float acc = 0.f;
  for (int m = threadIdx.x; m < M_; m += 256) {
    float p = c[m] * (1.0f / (float)B_);
    acc += p * logf(p + 1e-10f);
  }
  red[threadIdx.x] = acc; __syncthreads();
  for (int t = 128; t; t >>= 1) {
    if (threadIdx.x < t) red[threadIdx.x] += red[threadIdx.x + t];
    __syncthreads();
  }
  if (threadIdx.x == 0) out_perp[(ks >> 1) * 2 + (ks & 1)] = expf(-red[0]);
}

// ---------------------------------------------------------------------------
// cmcm softmax over M for stage K-1: logits row lives entirely in LDS (32KB).
// Emits ph (f16) and log(ph+1e-10) (f16) for the Scode WMMA GEMM.
__global__ __launch_bounds__(256) void softmax_kernel(
    const float* __restrict__ pcf, const float* __restrict__ plm,
    const float* __restrict__ emb, const float* __restrict__ enorm,
    _Float16* __restrict__ ph, _Float16* __restrict__ lph) {
  __shared__ float xs[D_];
  __shared__ float lg[M_];          // 32 KB logits row
  __shared__ float red[256];
  const int b = blockIdx.x, s = blockIdx.y;
  const float* x = (s ? plm : pcf) + (size_t)b * D_;
  for (int d = threadIdx.x; d < D_; d += 256) xs[d] = x[d];
  __syncthreads();
  float xn = 0.f;
  for (int d = 0; d < D_; ++d) xn += xs[d] * xs[d];
  const float* E3 = emb + (size_t)3 * M_ * D_;
  const float* en3 = enorm + (size_t)3 * M_;
  float mx = -3.402823466e38f;
  for (int m = threadIdx.x; m < M_; m += 256) {
    const float* er = E3 + (size_t)m * D_;
    float dot = 0.f;
    for (int d = 0; d < D_; ++d) dot = fmaf(xs[d], er[d], dot);
    float l = -sqrtf(fmaxf(en3[m] + xn - 2.0f * dot, 0.0f));
    lg[m] = l;
    mx = fmaxf(mx, l);
  }
  red[threadIdx.x] = mx; __syncthreads();
  for (int t = 128; t; t >>= 1) {
    if (threadIdx.x < t) red[threadIdx.x] = fmaxf(red[threadIdx.x], red[threadIdx.x + t]);
    __syncthreads();
  }
  mx = red[0]; __syncthreads();
  float sm = 0.f;
  for (int m = threadIdx.x; m < M_; m += 256) sm += expf(lg[m] - mx);
  red[threadIdx.x] = sm; __syncthreads();
  for (int t = 128; t; t >>= 1) {
    if (threadIdx.x < t) red[threadIdx.x] += red[threadIdx.x + t];
    __syncthreads();
  }
  const float inv = 1.0f / red[0];
  _Float16* php  = ph  + ((size_t)s * B_ + b) * M_;
  _Float16* lphp = lph + ((size_t)s * B_ + b) * M_;
  for (int m = threadIdx.x; m < M_; m += 256) {
    float p = expf(lg[m] - mx) * inv;
    php[m]  = (_Float16)p;
    lphp[m] = (_Float16)logf(p + 1e-10f);
  }
}

// ---------------------------------------------------------------------------
// Scode[i,j] = sum_m ph_pcf[i,m]*lph_plm[j,m] + ph_plm[i,m]*lph_pcf[j,m]
// [B,B,M] GEMM-NT via v_wmma_f32_16x16x32_f16 — the dominant ~550 GFLOP.
// Fragment quad (a1,b1,a2,b2) ping-pong double-buffered across the m-loop.
__device__ __forceinline__ void scode_load(HF16* __restrict__ f,
    const _Float16* __restrict__ ph_p, const _Float16* __restrict__ lph_l,
    const _Float16* __restrict__ ph_l, const _Float16* __restrict__ lph_p,
    size_t rowA, size_t rowB, int m0, int half) {
  f[0].h[0] = *(const v8h*)(ph_p  + rowA + m0 + half * 8);
  f[0].h[1] = *(const v8h*)(ph_p  + rowA + m0 + 16 + half * 8);
  f[1].h[0] = *(const v8h*)(lph_l + rowB + m0 + half * 8);
  f[1].h[1] = *(const v8h*)(lph_l + rowB + m0 + 16 + half * 8);
  f[2].h[0] = *(const v8h*)(ph_l  + rowA + m0 + half * 8);
  f[2].h[1] = *(const v8h*)(ph_l  + rowA + m0 + 16 + half * 8);
  f[3].h[0] = *(const v8h*)(lph_p + rowB + m0 + half * 8);
  f[3].h[1] = *(const v8h*)(lph_p + rowB + m0 + 16 + half * 8);
}

__global__ __launch_bounds__(128) void scode_kernel(
    const _Float16* __restrict__ ph, const _Float16* __restrict__ lph,
    float* __restrict__ scode) {
  const int tid = threadIdx.x, lane = tid & 31, wv = tid >> 5;
  const int half = lane >> 4, lr = lane & 15;
  const int i0 = blockIdx.y * 16;
  const int j0 = blockIdx.x * 64 + wv * 16;
  const _Float16* ph_p  = ph;
  const _Float16* ph_l  = ph  + (size_t)B_ * M_;
  const _Float16* lph_p = lph;
  const _Float16* lph_l = lph + (size_t)B_ * M_;
  const size_t rowA = (size_t)(i0 + lr) * M_;
  const size_t rowB = (size_t)(j0 + lr) * M_;

  HF16 f0[4], f1[4];
  scode_load(f0, ph_p, lph_l, ph_l, lph_p, rowA, rowB, 0, half);

  v8f c = {};
#pragma unroll 1
  for (int it = 0; it < M_ / 64; ++it) {
    const int m1 = it * 64 + 32;                    // always < M_
    __builtin_prefetch(ph_p + rowA + m1 + 512, 0, 1);   // global_prefetch_b8
    __builtin_prefetch(lph_l + rowB + m1 + 512, 0, 1);
    scode_load(f1, ph_p, lph_l, ph_l, lph_p, rowA, rowB, m1, half);
    c = __builtin_amdgcn_wmma_f32_16x16x32_f16(false, f0[0].v, false, f0[1].v, (short)0, c, false, false);
    c = __builtin_amdgcn_wmma_f32_16x16x32_f16(false, f0[2].v, false, f0[3].v, (short)0, c, false, false);
    const int m2 = m1 + 32;
    if (it + 1 < M_ / 64)
      scode_load(f0, ph_p, lph_l, ph_l, lph_p, rowA, rowB, m2, half);
    c = __builtin_amdgcn_wmma_f32_16x16x32_f16(false, f1[0].v, false, f1[1].v, (short)0, c, false, false);
    c = __builtin_amdgcn_wmma_f32_16x16x32_f16(false, f1[2].v, false, f1[3].v, (short)0, c, false, false);
  }
#pragma unroll
  for (int r = 0; r < 8; ++r)
    scode[(size_t)(i0 + r + 8 * half) * B_ + j0 + lr] = c[r];
}

// ---------------------------------------------------------------------------
__global__ void maxred_kernel(const float* __restrict__ scode, unsigned* __restrict__ menc) {
  __shared__ unsigned red[256];
  const size_t n = (size_t)B_ * B_;
  float mx = -3.402823466e38f;
  for (size_t i = (size_t)blockIdx.x * 256 + threadIdx.x; i < n; i += (size_t)gridDim.x * 256)
    mx = fmaxf(mx, -scode[i]);
  red[threadIdx.x] = fenc(mx); __syncthreads();
  for (int t = 128; t; t >>= 1) {
    if (threadIdx.x < t)
      red[threadIdx.x] = red[threadIdx.x] > red[threadIdx.x + t] ? red[threadIdx.x] : red[threadIdx.x + t];
    __syncthreads();
  }
  if (threadIdx.x == 0) atomicMax(menc, red[0]);
}

__global__ void ratio_kernel(const float* __restrict__ scode, const unsigned* __restrict__ menc,
                             float* __restrict__ lsum) {
  __shared__ float red[256];
  __shared__ float sdiag;
  const int i = blockIdx.x;
  const float C = fdec(*menc);     // MaxScode
  const float* row = scode + (size_t)i * B_;
  float sm = 0.f;
  for (int j = threadIdx.x; j < B_; j += 256) {
    float e = expf(row[j] + C);
    sm += e;
    if (j == i) sdiag = e;
  }
  red[threadIdx.x] = sm; __syncthreads();
  for (int t = 128; t; t >>= 1) {
    if (threadIdx.x < t) red[threadIdx.x] += red[threadIdx.x + t];
    __syncthreads();
  }
  if (threadIdx.x == 0) atomicAdd(lsum, logf(sdiag / (red[0] + 1e-5f)));
}

// commit-loss sums: acc[0]=Σ(pcf-q_p)^2 acc[1]=Σ(pcf-q_l)^2 acc[2]=Σ(plm-q_l)^2 acc[3]=Σ(plm-q_p)^2
__global__ void commit_kernel(const float* __restrict__ pcf, const float* __restrict__ plm,
                              const float* __restrict__ qp, const float* __restrict__ ql,
                              float* __restrict__ acc) {
  __shared__ float r0[256], r1[256], r2[256], r3[256];
  float a0 = 0, a1 = 0, a2 = 0, a3 = 0;
  const size_t n = (size_t)B_ * D_;
  for (size_t i = (size_t)blockIdx.x * 256 + threadIdx.x; i < n; i += (size_t)gridDim.x * 256) {
    float x = pcf[i], y = plm[i], u = qp[i], v = ql[i];
    float d0 = x - u, d1 = x - v, d2 = y - v, d3 = y - u;
    a0 += d0 * d0; a1 += d1 * d1; a2 += d2 * d2; a3 += d3 * d3;
  }
  r0[threadIdx.x] = a0; r1[threadIdx.x] = a1; r2[threadIdx.x] = a2; r3[threadIdx.x] = a3;
  __syncthreads();
  for (int t = 128; t; t >>= 1) {
    if (threadIdx.x < t) {
      r0[threadIdx.x] += r0[threadIdx.x + t]; r1[threadIdx.x] += r1[threadIdx.x + t];
      r2[threadIdx.x] += r2[threadIdx.x + t]; r3[threadIdx.x] += r3[threadIdx.x + t];
    }
    __syncthreads();
  }
  if (threadIdx.x == 0) {
    atomicAdd(&acc[0], r0[0]); atomicAdd(&acc[1], r1[0]);
    atomicAdd(&acc[2], r2[0]); atomicAdd(&acc[3], r3[0]);
  }
}

__global__ void finalize_kernel(const float* __restrict__ scal, float* __restrict__ out_loss) {
  const float invBD = 1.0f / ((float)B_ * (float)D_);
  float cm = 0.5f * (-scal[0] / (float)B_);
  float commit = (0.5f * scal[1] + 0.25f * scal[2] + 0.5f * scal[3] + 0.25f * scal[4]) * invBD;
  out_loss[0] = cm + commit;
}

__global__ void qcopy_kernel(const float* __restrict__ qp, const float* __restrict__ ql,
                             float* __restrict__ out) {
  size_t i = (size_t)blockIdx.x * 256 + threadIdx.x;
  const size_t n = (size_t)B_ * D_;
  if (i < n) { out[i] = qp[i]; out[n + i] = ql[i]; }
}

// ---------------------------------------------------------------------------
extern "C" void kernel_launch(void* const* d_in, const int* in_sizes, int n_in,
                              void* d_out, int out_size, void* d_ws, size_t ws_size,
                              hipStream_t stream) {
  (void)in_sizes; (void)n_in; (void)out_size; (void)ws_size;
  const float* pcf = (const float*)d_in[0];
  const float* plm = (const float*)d_in[1];
  const float* emb = (const float*)d_in[2];
  float* out = (float*)d_out;

  char* ws = (char*)d_ws;
  size_t off = 0;
  auto alloc = [&](size_t bytes) -> char* {
    char* p = ws + off;
    off = (off + bytes + 255) & ~(size_t)255;
    return p;
  };
  const size_t BD = (size_t)B_ * D_, BM = (size_t)B_ * M_;
  float*    res_p  = (float*)alloc(BD * 4);
  float*    res_l  = (float*)alloc(BD * 4);
  float*    q_p    = (float*)alloc(BD * 4);
  float*    q_l    = (float*)alloc(BD * 4);
  _Float16* resh_p = (_Float16*)alloc(BD * 2);
  _Float16* resh_l = (_Float16*)alloc(BD * 2);
  _Float16* Eh     = (_Float16*)alloc((size_t)K_ * M_ * D_ * 2);
  float*    enorm  = (float*)alloc((size_t)K_ * M_ * 4);
  int*      amin_p = (int*)alloc((size_t)B_ * 4);
  int*      amin_l = (int*)alloc((size_t)B_ * 4);
  float*    counts = (float*)alloc((size_t)K_ * 2 * M_ * 4);
  float*    scal   = (float*)alloc(64 * 4);
  _Float16* ph     = (_Float16*)alloc(2 * BM * 2);   // 134 MB
  _Float16* lph    = (_Float16*)alloc(2 * BM * 2);   // 134 MB
  float*    scode  = (float*)alloc((size_t)B_ * B_ * 4); // 67 MB

  float* out_q    = out;
  float* out_loss = out + 2 * BD;
  float* out_sem  = out + 2 * BD + 1;
  float* out_perp = out + 2 * BD + 1 + (size_t)2 * B_ * K_;

  init_main_kernel<<<(unsigned)((BD + 255) / 256), 256, 0, stream>>>(
      pcf, plm, res_p, res_l, resh_p, resh_l, q_p, q_l);
  init_misc_kernel<<<(unsigned)(((size_t)K_ * 2 * M_ + 255) / 256), 256, 0, stream>>>(counts, scal);
  eprep_kernel<<<(K_ * M_) / 8, 256, 0, stream>>>(emb, Eh, enorm);

  for (int k = 0; k < K_; ++k) {
    for (int s = 0; s < 2; ++s) {
      argmin_kernel<<<B_ / 16, 128, 0, stream>>>(
          s ? resh_l : resh_p, Eh, enorm, k, s ? amin_l : amin_p);
      update_kernel<<<B_ / 8, 256, 0, stream>>>(
          emb, s ? amin_l : amin_p,
          s ? res_l : res_p, s ? resh_l : resh_p, s ? q_l : q_p,
          counts + ((size_t)k * 2 + s) * M_,
          out_sem + (size_t)s * B_ * K_, k);
    }
  }

  perp_kernel<<<K_ * 2, 256, 0, stream>>>(counts, out_perp);
  softmax_kernel<<<dim3(B_, 2), 256, 0, stream>>>(pcf, plm, emb, enorm, ph, lph);
  scode_kernel<<<dim3(B_ / 64, B_ / 16), 128, 0, stream>>>(ph, lph, scode);
  maxred_kernel<<<4096, 256, 0, stream>>>(scode, (unsigned*)(scal + 8));
  ratio_kernel<<<B_, 256, 0, stream>>>(scode, (const unsigned*)(scal + 8), scal + 0);
  commit_kernel<<<1024, 256, 0, stream>>>(pcf, plm, q_p, q_l, scal + 1);
  finalize_kernel<<<1, 1, 0, stream>>>(scal, out_loss);
  qcopy_kernel<<<(unsigned)((BD + 255) / 256), 256, 0, stream>>>(q_p, q_l, out_q);
}
